// ContextAttention_38233798869160
// MI455X (gfx1250) — matrix operand order repaired and checked by reading, not compiled, not run on hardware
//
#include <hip/hip_runtime.h>
#include <hip/hip_bf16.h>

// ---------------------------------------------------------------------------
// Types
// ---------------------------------------------------------------------------
typedef __bf16 bf16;
typedef __attribute__((ext_vector_type(16))) __bf16 v16bf;
typedef __attribute__((ext_vector_type(8)))  float  v8f;
typedef __attribute__((ext_vector_type(4)))  unsigned int u32x4;

union Frag { v16bf v; u32x4 q[2]; };

// ---------------------------------------------------------------------------
// Constants (match reference)
// ---------------------------------------------------------------------------
#define BATCH   4
#define SEQ     4096
#define CH      768
#define NHEAD   12
#define HD      64
#define M_ROWS  (BATCH * SEQ)     // 16384
#define C3      (3 * CH)          // 2304
#define SCALE   0.125f            // 64^-0.5
#define EPS_F   1e-10f
#define DELU_P_F 10.0f

__device__ __forceinline__ float delu_f(float x) {
    float pos = fmaxf(x, 0.0f);
    float neg = fminf(x, 0.0f);
    return DELU_P_F * pos + __expf(DELU_P_F * neg);
}

__device__ __forceinline__ bf16 f2bf(float f) {
    unsigned u = __builtin_bit_cast(unsigned, f);
    u += 0x7FFFu + ((u >> 16) & 1u);   // round-to-nearest-even
    unsigned short s = (unsigned short)(u >> 16);
    return __builtin_bit_cast(bf16, s);
}

__device__ __forceinline__ float bf2f(bf16 b) {
    unsigned short s = __builtin_bit_cast(unsigned short, b);
    unsigned u = ((unsigned)s) << 16;
    return __builtin_bit_cast(float, u);
}

// LDS byte offset from a generic (flat) pointer into __shared__:
// flat LDS aperture maps to LDS via addr[31:0] (ISA aperture rules).
__device__ __forceinline__ unsigned lds_off(const void* p) {
    return (unsigned)(unsigned long long)(uintptr_t)p;
}

// ---------------------------------------------------------------------------
// Kernel: f32 -> bf16 convert (grid-stride)
// ---------------------------------------------------------------------------
__global__ void cvt_f32_bf16(const float* __restrict__ src,
                             bf16* __restrict__ dst, int n) {
    int i = blockIdx.x * blockDim.x + threadIdx.x;
    int stride = gridDim.x * blockDim.x;
    for (; i < n; i += stride) dst[i] = f2bf(src[i]);
}

// ---------------------------------------------------------------------------
// Kernel: init diag accumulators
// ---------------------------------------------------------------------------
__global__ void init_diag(float* __restrict__ kv, float* __restrict__ lkv,
                          float* __restrict__ ks, int n) {
    int i = blockIdx.x * blockDim.x + threadIdx.x;
    if (i < n) { kv[i] = 0.0f; lkv[i] = 0.0f; ks[i] = EPS_F; }
}

// ---------------------------------------------------------------------------
// WMMA GEMM:  out[M x N] = A[M x K] * B[N x K]^T   (bf16 in, f32 acc)
//  mode 0: qkv epilogue -> delu on cols < 2*CH, store bf16 to outB
//  mode 1: projection epilogue -> + bias, store f32 to outF
// Block tile 128x128, BK=32, 256 threads = 8 waves (2 M x 4 N),
// wave tile 64x32 = 4x2 WMMA 16x16 tiles.
// Double-buffered LDS fed by GLOBAL_LOAD_ASYNC_TO_LDS_B128 (ASYNCcnt).
// ---------------------------------------------------------------------------
#define BM 128
#define BN 128
#define BK 32

__global__ __launch_bounds__(256)
void gemm_bf16_wmma(const bf16* __restrict__ A, const bf16* __restrict__ B,
                    const float* __restrict__ bias,
                    bf16* __restrict__ outB, float* __restrict__ outF,
                    int N, int K, int mode) {
    __shared__ bf16 As[2][BM][BK];
    __shared__ bf16 Bs[2][BN][BK];

    const int tid  = threadIdx.x;
    const int lane = tid & 31;
    const int wid  = tid >> 5;
    const int wm   = wid >> 2;      // 0..1
    const int wn   = wid & 3;       // 0..3
    const int lrow = lane & 15;
    const int lhi  = lane >> 4;     // 0 or 1

    const int m0 = blockIdx.x * BM;
    const int n0 = blockIdx.y * BN;

    v8f acc[4][2] = {};             // 64 VGPRs of f32 accumulators

    const int r  = tid >> 1;        // 0..127 : tile row loaded by this thread
    const int kc = (tid & 1) << 4;  // 0 or 16

    const bf16* gA = A + (size_t)(m0 + r) * K + kc;
    const bf16* gB = B + (size_t)(n0 + r) * K + kc;

    // Per-thread LDS destinations for the two buffers (byte offsets).
    const unsigned la[2] = { lds_off(&As[0][r][kc]), lds_off(&As[1][r][kc]) };
    const unsigned lb[2] = { lds_off(&Bs[0][r][kc]), lds_off(&Bs[1][r][kc]) };

    // Issue one K-tile into LDS buffer `buf` (4 async b128 loads / thread).
    // INST_OFFSET is added to BOTH the LDS and the global address, so one
    // (lds, global) pair covers both 16B chunks via offset:0 / offset:16.
    auto issue_tile = [&](int buf, int k0) {
        const bf16* ga = gA + k0;
        const bf16* gb = gB + k0;
        asm volatile(
            "global_load_async_to_lds_b128 %0, %1, off\n\t"
            "global_load_async_to_lds_b128 %0, %1, off offset:16\n\t"
            "global_load_async_to_lds_b128 %2, %3, off\n\t"
            "global_load_async_to_lds_b128 %2, %3, off offset:16"
            :
            : "v"(la[buf]), "v"(ga), "v"(lb[buf]), "v"(gb)
            : "memory");
    };

    const int nIter = K / BK;
    issue_tile(0, 0);

    for (int it = 0; it < nIter; ++it) {
        const int cur = it & 1;

        // My async loads for buffer `cur` have landed in LDS.
        asm volatile("s_wait_asynccnt 0x0" ::: "memory");
        // All waves' loads landed; all waves done reading buffer cur^1.
        __syncthreads();

        if (it + 1 < nIter) issue_tile(cur ^ 1, (it + 1) * BK);

        // B fragments: lane<16 holds K=0..15 of col lane%16, lane>=16 K=16..31
        Frag fb[2];
        #pragma unroll
        for (int ni = 0; ni < 2; ni++) {
            int col = wn * 32 + ni * 16 + lrow;
            fb[ni].q[0] = *(const u32x4*)&Bs[cur][col][lhi * 16];
            fb[ni].q[1] = *(const u32x4*)&Bs[cur][col][lhi * 16 + 8];
        }
        #pragma unroll
        for (int mi = 0; mi < 4; mi++) {
            // A fragment: lane<16 K=0..15, lane>=16 K=16..31
            int row = wm * 64 + mi * 16 + lrow;
            Frag fa;
            fa.q[0] = *(const u32x4*)&As[cur][row][lhi * 16];
            fa.q[1] = *(const u32x4*)&As[cur][row][lhi * 16 + 8];
            #pragma unroll
            for (int ni = 0; ni < 2; ni++) {
                acc[mi][ni] = __builtin_amdgcn_wmma_f32_16x16x32_bf16(
                    false, fa.v, false, fb[ni].v, (short)0, acc[mi][ni],
                    false, false);
            }
        }
    }

    // Epilogue. C/D layout: lane col = lane%16; VGPR rr -> row rr (+8 if lane>=16)
    #pragma unroll
    for (int mi = 0; mi < 4; mi++) {
        #pragma unroll
        for (int ni = 0; ni < 2; ni++) {
            int col  = n0 + wn * 32 + ni * 16 + lrow;
            int rowb = m0 + wm * 64 + mi * 16 + lhi * 8;
            #pragma unroll
            for (int rr = 0; rr < 8; rr++) {
                float v = acc[mi][ni][rr];
                size_t idx = (size_t)(rowb + rr) * N + col;
                if (mode == 0) {
                    if (col < 2 * CH) v = delu_f(v);   // q and k thirds
                    outB[idx] = f2bf(v);
                } else {
                    outF[idx] = v + bias[col];
                }
            }
        }
    }
}

// ---------------------------------------------------------------------------
// Reductions over N:  kv_diag, l_kv_diag, k_sum   (per (b, c), c = h*64+d)
// grid (C/256, B, SEQ/256), block 256. Coalesced column access per row.
// ---------------------------------------------------------------------------
__global__ __launch_bounds__(256)
void reduce_diag(const bf16* __restrict__ qkvb, const float* __restrict__ y,
                 float* __restrict__ kv, float* __restrict__ lkv,
                 float* __restrict__ ks) {
    int c  = blockIdx.x * 256 + threadIdx.x;   // 0..767
    int b  = blockIdx.y;
    int n0 = blockIdx.z * 256;

    float skv = 0.0f, slkv = 0.0f, sk = 0.0f;
    for (int i = 0; i < 256; i++) {
        size_t row = (size_t)(b * SEQ + n0 + i);
        float kd = bf2f(qkvb[row * C3 + CH + c]);        // delu already applied
        float vv = bf2f(qkvb[row * C3 + 2 * CH + c]);    // raw v
        float lk = delu_f(y[row * CH + c]);
        skv  += kd * vv;
        slkv += lk * vv;
        sk   += kd;
    }
    atomicAdd(&kv [b * CH + c], skv  * SCALE);
    atomicAdd(&lkv[b * CH + c], slkv * SCALE);
    atomicAdd(&ks [b * CH + c], sk);
}

// ---------------------------------------------------------------------------
// Norm + attention vectors.  grid (M_ROWS, NHEAD), block 64 (= one head).
// norm[b,h,n] = 1 / sum_d q[b,h,n,d]*k_sum[b,h,d]
// attnL = q * kv_diag * norm ; attnG = q * l_kv_diag * norm  (bf16 out)
// ---------------------------------------------------------------------------
__global__ __launch_bounds__(64)
void norm_attn(const bf16* __restrict__ qkvb,
               const float* __restrict__ kv, const float* __restrict__ lkv,
               const float* __restrict__ ks,
               bf16* __restrict__ attnL, bf16* __restrict__ attnG) {
    int row = blockIdx.x;
    int h   = blockIdx.y;
    int b   = row >> 12;               // /SEQ
    int t   = threadIdx.x;             // d in 0..63
    int c   = h * HD + t;

    float q = bf2f(qkvb[(size_t)row * C3 + c]);
    float dot = q * ks[b * CH + c];

    __shared__ float red[64];
    red[t] = dot;
    __syncthreads();
    for (int s = 32; s > 0; s >>= 1) {
        if (t < s) red[t] += red[t + s];
        __syncthreads();
    }
    float ninv = 1.0f / red[0];

    size_t oidx = (size_t)row * CH + c;
    attnL[oidx] = f2bf(q * kv [b * CH + c] * ninv);
    attnG[oidx] = f2bf(q * lkv[b * CH + c] * ninv);
}

// ---------------------------------------------------------------------------
// Launcher
// ---------------------------------------------------------------------------
extern "C" void kernel_launch(void* const* d_in, const int* in_sizes, int n_in,
                              void* d_out, int out_size, void* d_ws, size_t ws_size,
                              hipStream_t stream) {
    const float* x       = (const float*)d_in[0];
    const float* y       = (const float*)d_in[1];
    const float* qkv_w   = (const float*)d_in[2];
    const float* proj1_w = (const float*)d_in[3];
    const float* proj1_b = (const float*)d_in[4];
    const float* proj2_w = (const float*)d_in[5];
    const float* proj2_b = (const float*)d_in[6];
    float* out = (float*)d_out;

    // Workspace carve-out (256B aligned)
    char* base = (char*)d_ws;
    size_t off = 0;
    auto carve = [&](size_t bytes) -> char* {
        char* p = base + off;
        off = (off + bytes + 255) & ~(size_t)255;
        return p;
    };
    bf16*  xb    = (bf16*) carve((size_t)M_ROWS * CH * sizeof(bf16));
    bf16*  wqkv  = (bf16*) carve((size_t)C3 * CH * sizeof(bf16));
    bf16*  p1b   = (bf16*) carve((size_t)CH * CH * sizeof(bf16));
    bf16*  p2b   = (bf16*) carve((size_t)CH * CH * sizeof(bf16));
    bf16*  qkvb  = (bf16*) carve((size_t)M_ROWS * C3 * sizeof(bf16));
    float* kvD   = (float*)carve((size_t)BATCH * CH * sizeof(float));
    float* lkvD  = (float*)carve((size_t)BATCH * CH * sizeof(float));
    float* ksum  = (float*)carve((size_t)BATCH * CH * sizeof(float));
    bf16*  attnL = (bf16*) carve((size_t)M_ROWS * CH * sizeof(bf16));
    bf16*  attnG = (bf16*) carve((size_t)M_ROWS * CH * sizeof(bf16));

    // 1) bf16 converts
    cvt_f32_bf16<<<4096, 256, 0, stream>>>(x, xb, M_ROWS * CH);
    cvt_f32_bf16<<<2048, 256, 0, stream>>>(qkv_w, wqkv, C3 * CH);
    cvt_f32_bf16<<<1024, 256, 0, stream>>>(proj1_w, p1b, CH * CH);
    cvt_f32_bf16<<<1024, 256, 0, stream>>>(proj2_w, p2b, CH * CH);

    // 2) init diag accumulators
    init_diag<<<(BATCH * CH + 255) / 256, 256, 0, stream>>>(kvD, lkvD, ksum,
                                                            BATCH * CH);

    // 3) QKV GEMM (+delu on q,k) -> qkvb
    gemm_bf16_wmma<<<dim3(M_ROWS / BM, C3 / BN), 256, 0, stream>>>(
        xb, wqkv, nullptr, qkvb, nullptr, C3, CH, 0);

    // 4) diagonal reductions
    reduce_diag<<<dim3(CH / 256, BATCH, SEQ / 256), 256, 0, stream>>>(
        qkvb, y, kvD, lkvD, ksum);

    // 5) norm + attention vectors (bf16)
    norm_attn<<<dim3(M_ROWS, NHEAD), 64, 0, stream>>>(qkvb, kvD, lkvD, ksum,
                                                      attnL, attnG);

    // 6) projections -> d_out (xo then yo)
    gemm_bf16_wmma<<<dim3(M_ROWS / BM, CH / BN), 256, 0, stream>>>(
        attnL, p1b, proj1_b, nullptr, out, CH, CH, 1);
    gemm_bf16_wmma<<<dim3(M_ROWS / BM, CH / BN), 256, 0, stream>>>(
        attnG, p2b, proj2_b, nullptr, out + (size_t)M_ROWS * CH, CH, CH, 1);
}
